// BLIPConceptPrefixModelV3_61117384622864
// MI455X (gfx1250) — matrix operand order repaired and checked by reading, not compiled, not run on hardware
//
#include <hip/hip_runtime.h>

// ---------------------------------------------------------------------------
// Types for CDNA5 WMMA (wave32, 16x16x32 bf16 -> f32)
// ---------------------------------------------------------------------------
typedef __attribute__((ext_vector_type(16))) __bf16 v16bf;
typedef __attribute__((ext_vector_type(8)))  __bf16 v8bf;
typedef __attribute__((ext_vector_type(8)))  float  v8f;

#define BATCH 16
#define SFULL 577
#define SP    576
#define DIM   768
#define NCON  256
#define NCLS  1000
#define KTOP  16

__device__ __forceinline__ v8f bfmma(v16bf a, v16bf b, v8f c) {
    // D = A(16x32 bf16) x B(32x16 bf16) + C(16x16 f32)
    return __builtin_amdgcn_wmma_f32_16x16x32_bf16(
        /*neg_a=*/false, a, /*neg_b=*/false, b,
        /*c_mod=*/(short)0, c, /*reuse_a=*/false, /*reuse_b=*/false);
}

// Split an f32 into bf16 hi + bf16 residual (bf16x3 compensated product).
__device__ __forceinline__ void split_hilo(float x, __bf16& h, __bf16& l) {
    __bf16 hb = (__bf16)x;               // RNE f32->bf16 (v_cvt_pk_bf16_f32)
    h = hb;
    l = (__bf16)(x - (float)hb);         // residual fits bf16 dynamic range
}

// A-fragment from LDS: per ISA, lanes 0-15 hold K{0..7,16..23}, lanes 16-31
// hold K{8..15,24..31}; p points at (row*STRIDE + khalf) with khalf=(lane>>4)*8.
__device__ __forceinline__ v16bf frag_a_lds(const __bf16* p) {
    v8bf a = *(const v8bf*)(p);          // K base..base+7      (ds_load_b128)
    v8bf c = *(const v8bf*)(p + 16);     // K base+16..base+23  (ds_load_b128)
    return __builtin_shufflevector(a, c, 0,1,2,3,4,5,6,7,8,9,10,11,12,13,14,15);
}

// B-fragment from LDS: lanes 0-15 hold K{0..7,16..23}, lanes 16-31 hold
// K{8..15,24..31}; p points at (row*STRIDE + (lane>>4)*8).
__device__ __forceinline__ v16bf frag_b_lds(const __bf16* p) {
    v8bf a = *(const v8bf*)(p);
    v8bf c = *(const v8bf*)(p + 16);
    return __builtin_shufflevector(a, c, 0,1,2,3,4,5,6,7,8,9,10,11,12,13,14,15);
}

// ---------------------------------------------------------------------------
// Kernel 1: qk[b,c,s] = sum_d concept_w[c,d] * q[b,1+s,d]
// 128x64 output tile per workgroup, 2x2 WMMA tiles per wave, K-chunks of 32,
// bf16x3 compensation, register double-buffered global loads.
// ---------------------------------------------------------------------------
#define LDSTRIDE 40   // 32 data + 8 pad bf16 elements -> conflict-free b128 reads

__global__ __launch_bounds__(256) void qk_gemm_kernel(
    const float* __restrict__ q, const float* __restrict__ cw,
    float* __restrict__ qk)
{
    __shared__ __bf16 Ah[128 * LDSTRIDE];
    __shared__ __bf16 Al[128 * LDSTRIDE];
    __shared__ __bf16 Bh[64 * LDSTRIDE];
    __shared__ __bf16 Bl[64 * LDSTRIDE];

    const int m0   = blockIdx.x * 128;         // concept rows
    const int n0   = blockIdx.y * 64;          // patch cols (s)
    const int b    = blockIdx.z;
    const int tid  = threadIdx.x;
    const int lane = tid & 31;
    const int wave = tid >> 5;

    // 8 waves in a 4x2 grid; each wave owns a 2x2 block of 16x16 tiles.
    const int wr  = wave & 3;                  // M-block 0..3 (rows wr*32..+31)
    const int wc  = wave >> 2;                 // N-block 0..1 (cols wc*32..+31)
    const int mi0 = wr * 2, mi1 = mi0 + 1;
    const int ni0 = wc * 2, ni1 = ni0 + 1;

    const int l15    = lane & 15;
    const int khalfA = (lane >> 4) * 8;        // A fragment K offset
    const int kbaseB = (lane >> 4) * 8;        // B fragment K offset

    const float* qb = q + ((size_t)b * SFULL + 1 + n0) * DIM;  // skip CLS token

    // Per-thread staging pointers: A = 1024 float4 slots, B = 512 slots.
    const float* ga[4];
    int oa[4];
    #pragma unroll
    for (int r = 0; r < 4; ++r) {
        int i   = tid + r * 256;               // 0..1023
        int row = i >> 3;
        int col = (i & 7) * 4;
        ga[r] = cw + (size_t)(m0 + row) * DIM + col;
        oa[r] = row * LDSTRIDE + col;
    }
    const float* gb[2];
    int ob[2];
    #pragma unroll
    for (int r = 0; r < 2; ++r) {
        int i   = tid + r * 256;               // 0..511
        int row = i >> 3;
        int col = (i & 7) * 4;
        gb[r] = qb + (size_t)row * DIM + col;
        ob[r] = row * LDSTRIDE + col;
    }

    v8f acc00 = {}, acc01 = {}, acc10 = {}, acc11 = {};

    // Preload K-chunk 0 into registers.
    float4 ra[4], rb[2];
    #pragma unroll
    for (int r = 0; r < 4; ++r) ra[r] = *(const float4*)(ga[r]);
    #pragma unroll
    for (int r = 0; r < 2; ++r) rb[r] = *(const float4*)(gb[r]);

    for (int kc = 0; kc < DIM; kc += 32) {
        // ---- convert staged registers to bf16 hi/lo and store to LDS ----
        #pragma unroll
        for (int r = 0; r < 4; ++r) {
            float f[4] = {ra[r].x, ra[r].y, ra[r].z, ra[r].w};
            #pragma unroll
            for (int j = 0; j < 4; ++j) split_hilo(f[j], Ah[oa[r] + j], Al[oa[r] + j]);
        }
        #pragma unroll
        for (int r = 0; r < 2; ++r) {
            float f[4] = {rb[r].x, rb[r].y, rb[r].z, rb[r].w};
            #pragma unroll
            for (int j = 0; j < 4; ++j) split_hilo(f[j], Bh[ob[r] + j], Bl[ob[r] + j]);
        }
        __syncthreads();

        // ---- double-buffer: issue next chunk's global loads now, so their
        //      latency overlaps the 12 WMMAs below ----
        if (kc + 32 < DIM) {
            #pragma unroll
            for (int r = 0; r < 4; ++r) ra[r] = *(const float4*)(ga[r] + kc + 32);
            #pragma unroll
            for (int r = 0; r < 2; ++r) rb[r] = *(const float4*)(gb[r] + kc + 32);
        }

        // ---- compute: bf16x3 compensated WMMA, 2x2 tiles ----
        v16bf ah0 = frag_a_lds(&Ah[(mi0 * 16 + l15) * LDSTRIDE + khalfA]);
        v16bf al0 = frag_a_lds(&Al[(mi0 * 16 + l15) * LDSTRIDE + khalfA]);
        v16bf ah1 = frag_a_lds(&Ah[(mi1 * 16 + l15) * LDSTRIDE + khalfA]);
        v16bf al1 = frag_a_lds(&Al[(mi1 * 16 + l15) * LDSTRIDE + khalfA]);
        v16bf bh0 = frag_b_lds(&Bh[(ni0 * 16 + l15) * LDSTRIDE + kbaseB]);
        v16bf bl0 = frag_b_lds(&Bl[(ni0 * 16 + l15) * LDSTRIDE + kbaseB]);
        v16bf bh1 = frag_b_lds(&Bh[(ni1 * 16 + l15) * LDSTRIDE + kbaseB]);
        v16bf bl1 = frag_b_lds(&Bl[(ni1 * 16 + l15) * LDSTRIDE + kbaseB]);

        acc00 = bfmma(ah0, bh0, acc00);
        acc00 = bfmma(ah0, bl0, acc00);
        acc00 = bfmma(al0, bh0, acc00);

        acc01 = bfmma(ah0, bh1, acc01);
        acc01 = bfmma(ah0, bl1, acc01);
        acc01 = bfmma(al0, bh1, acc01);

        acc10 = bfmma(ah1, bh0, acc10);
        acc10 = bfmma(ah1, bl0, acc10);
        acc10 = bfmma(al1, bh0, acc10);

        acc11 = bfmma(ah1, bh1, acc11);
        acc11 = bfmma(ah1, bl1, acc11);
        acc11 = bfmma(al1, bh1, acc11);

        __syncthreads();
    }

    // ---- epilogue: C/D layout (VGPR j: M=j / j+8 by lane half) ----
    #pragma unroll
    for (int j = 0; j < 8; ++j) {
        int mr0 = m0 + mi0 * 16 + j + 8 * (lane >> 4);
        int mr1 = m0 + mi1 * 16 + j + 8 * (lane >> 4);
        size_t base0 = ((size_t)b * NCON + mr0) * SP + n0;
        size_t base1 = ((size_t)b * NCON + mr1) * SP + n0;
        qk[base0 + ni0 * 16 + l15] = acc00[j];
        qk[base0 + ni1 * 16 + l15] = acc01[j];
        qk[base1 + ni0 * 16 + l15] = acc10[j];
        qk[base1 + ni1 * 16 + l15] = acc11[j];
    }
}

// ---------------------------------------------------------------------------
// Kernel 2: per (b,c) row: top-16 of 576 (tie -> lower index), softmax weights
// One wave per row, 4 waves per block.
// ---------------------------------------------------------------------------
__global__ __launch_bounds__(128) void topk_softmax_kernel(
    const float* __restrict__ qk, float* __restrict__ wval, int* __restrict__ widx)
{
    __shared__ float buf[4 * SP];
    const int wave = threadIdx.x >> 5;
    const int lane = threadIdx.x & 31;
    const int row  = blockIdx.x * 4 + wave;       // 0..B*NCON-1

    const float* src = qk + (size_t)row * SP;
    float* bp = buf + wave * SP;
    #pragma unroll
    for (int i = 0; i < SP / 32; ++i) bp[lane + 32 * i] = src[lane + 32 * i];
    __builtin_amdgcn_wave_barrier();

    float tv[KTOP];
    int   ti[KTOP];
    #pragma unroll
    for (int it = 0; it < KTOP; ++it) {
        float mv = -__builtin_inff();
        int   mi = 0x7fffffff;
        #pragma unroll
        for (int i = 0; i < SP / 32; ++i) {
            int s = lane + 32 * i;
            float v = bp[s];
            if (v > mv || (v == mv && s < mi)) { mv = v; mi = s; }
        }
        #pragma unroll
        for (int off = 16; off > 0; off >>= 1) {
            float ov = __shfl_xor(mv, off, 32);
            int   oi = __shfl_xor(mi, off, 32);
            if (ov > mv || (ov == mv && oi < mi)) { mv = ov; mi = oi; }
        }
        if (lane == 0) bp[mi] = -__builtin_inff();
        __builtin_amdgcn_wave_barrier();
        tv[it] = mv;
        ti[it] = mi;
    }

    if (lane == 0) {
        float mx = tv[0];                  // first pick is the max
        float e[KTOP], s = 0.f;
        #pragma unroll
        for (int i = 0; i < KTOP; ++i) { e[i] = __expf(tv[i] - mx); s += e[i]; }
        float inv = 1.f / s;
        #pragma unroll
        for (int i = 0; i < KTOP; ++i) {
            wval[(size_t)row * KTOP + i] = e[i] * inv;
            widx[(size_t)row * KTOP + i] = ti[i];
        }
    }
}

// ---------------------------------------------------------------------------
// Kernel 3: deterministic weighted gather: partial h over c-chunks of 64
// hpart[cch][b][d] = sum over 1024 (c,k) entries of w * q[b, 1+idx, d]
// ---------------------------------------------------------------------------
__global__ __launch_bounds__(256) void h_partial_kernel(
    const float* __restrict__ q, const int* __restrict__ widx,
    const float* __restrict__ wval, float* __restrict__ hpart)
{
    __shared__ int   sidx[64 * KTOP];
    __shared__ float swv [64 * KTOP];
    const int d   = blockIdx.x * 256 + threadIdx.x;  // 0..767
    const int cch = blockIdx.y;                      // 0..3
    const int b   = blockIdx.z;

    const int base = (b * NCON + cch * 64) * KTOP;   // 1024 entries
    for (int i = threadIdx.x; i < 64 * KTOP; i += 256) {
        sidx[i] = widx[base + i];
        swv[i]  = wval[base + i];
    }
    __syncthreads();

    const float* qb = q + ((size_t)b * SFULL + 1) * DIM + d;
    float acc = 0.f;
    #pragma unroll 4
    for (int j = 0; j < 64 * KTOP; ++j)              // fixed order -> deterministic
        acc += swv[j] * qb[(size_t)sidx[j] * DIM];

    hpart[((size_t)cch * BATCH + b) * DIM + d] = acc;
}

// ---------------------------------------------------------------------------
// Kernel 4: reduce 4 partials (fixed order), mean over C, ReLU
// ---------------------------------------------------------------------------
__global__ __launch_bounds__(256) void h_reduce_kernel(
    const float* __restrict__ hpart, float* __restrict__ hmat)
{
    const int i = blockIdx.x * 256 + threadIdx.x;    // 0..BATCH*DIM-1
    const int n = BATCH * DIM;
    float s = ((hpart[i] + hpart[n + i]) + (hpart[2 * n + i] + hpart[3 * n + i]))
              * (1.0f / (float)NCON);
    hmat[i] = s > 0.f ? s : 0.f;                     // ReLU fused
}

// ---------------------------------------------------------------------------
// Kernel 5: y = relu(h)(16x768) @ cls_w^T(768x1000) + cls_b, WMMA bf16x3
// Fragments built directly from L2-resident f32 (no LDS needed).
// ---------------------------------------------------------------------------
__device__ __forceinline__ void frag_a_f32(const float* p, int khalf,
                                           v16bf& fh, v16bf& fl) {
    v16bf h, l;
    #pragma unroll
    for (int i = 0; i < 8; ++i) {
        __bf16 h0, l0, h1, l1;
        split_hilo(p[khalf + i],      h0, l0);
        split_hilo(p[khalf + 8 + i],  h1, l1);
        h[i] = h0; h[i + 8] = h1;
        l[i] = l0; l[i + 8] = l1;
    }
    fh = h; fl = l;
}

__device__ __forceinline__ void frag_b_f32(const float* p, int kbase, bool valid,
                                           v16bf& fh, v16bf& fl) {
    v16bf h, l;
    #pragma unroll
    for (int i = 0; i < 16; ++i) {
        float x = valid ? p[kbase + i] : 0.f;
        __bf16 hb, lb;
        split_hilo(x, hb, lb);
        h[i] = hb; l[i] = lb;
    }
    fh = h; fl = l;
}

__global__ __launch_bounds__(256) void cls_gemm_kernel(
    const float* __restrict__ hmat, const float* __restrict__ clw,
    const float* __restrict__ clb, float* __restrict__ y)
{
    const int wave = threadIdx.x >> 5;
    const int lane = threadIdx.x & 31;
    const int gw   = blockIdx.x * 8 + wave;          // 0..63 N-tiles
    const int n0   = gw * 16;
    const int l15  = lane & 15;
    const int nl   = n0 + l15;
    const bool valid = (nl < NCLS);

    const int khalfA = (lane >> 4) * 16;
    const int kbaseB = (lane >> 4) * 16;

    const float* arow = hmat + (size_t)l15 * DIM;                 // M = b row
    const float* brow = clw + (size_t)(valid ? nl : 0) * DIM;     // safe row

    v8f acc = {};
    for (int kb = 0; kb < DIM; kb += 32) {
        v16bf ah, al, bh, bl;
        frag_a_f32(arow + kb, khalfA, ah, al);
        frag_b_f32(brow + kb, kbaseB, valid, bh, bl);
        acc = bfmma(ah, bh, acc);
        acc = bfmma(ah, bl, acc);
        acc = bfmma(al, bh, acc);
    }

    if (valid) {
        float bias = clb[nl];
        #pragma unroll
        for (int j = 0; j < 8; ++j) {
            int m = j + 8 * (lane >> 4);
            y[(size_t)m * NCLS + nl] = acc[j] + bias;
        }
    }
}

// ---------------------------------------------------------------------------
// Launch
// ---------------------------------------------------------------------------
extern "C" void kernel_launch(void* const* d_in, const int* in_sizes, int n_in,
                              void* d_out, int out_size, void* d_ws, size_t ws_size,
                              hipStream_t stream) {
    (void)in_sizes; (void)n_in; (void)out_size; (void)ws_size;

    const float* q   = (const float*)d_in[0];   // (16,577,768)
    const float* cw  = (const float*)d_in[1];   // (256,768)
    const float* clw = (const float*)d_in[2];   // (1000,768)
    const float* clb = (const float*)d_in[3];   // (1000,)
    float* y = (float*)d_out;                   // (16,1000)

    char* ws = (char*)d_ws;
    float* qk    = (float*)(ws);                          //  9,437,184 B
    float* wval  = (float*)(ws + 9437184);                //    262,144 B
    int*   widx  = (int*)  (ws + 9699328);                //    262,144 B
    float* hpart = (float*)(ws + 9961472);                //    196,608 B
    float* hmat  = (float*)(ws + 10158080);               //     49,152 B

    qk_gemm_kernel     <<<dim3(NCON / 128, SP / 64, BATCH), 256, 0, stream>>>(q, cw, qk);
    topk_softmax_kernel<<<dim3(BATCH * NCON / 4),           128, 0, stream>>>(qk, wval, widx);
    h_partial_kernel   <<<dim3(DIM / 256, 4, BATCH),        256, 0, stream>>>(q, widx, wval, hpart);
    h_reduce_kernel    <<<dim3(BATCH * DIM / 256),          256, 0, stream>>>(hpart, hmat);
    cls_gemm_kernel    <<<dim3(8),                          256, 0, stream>>>(hmat, clw, clb, y);
}
